// AttentionBlock_55027120996807
// MI455X (gfx1250) — matrix operand order repaired and checked by reading, not compiled, not run on hardware
//
#include <hip/hip_runtime.h>

// AttentionBlock for MI455X (gfx1250): GroupNorm -> QKV GEMM -> flash attention
// -> proj GEMM + residual. Matmuls via v_wmma_f32_16x16x32_bf16 (bf16 operands,
// fp32 accumulate). Wave32; 256-thread blocks = 8 waves. LDS operand gathers
// are contiguous ds_load_b128; straight global->LDS stages use the gfx1250
// async-to-LDS path (ASYNCcnt) via __builtin_amdgcn_global_load_async_to_lds_b128.

typedef __attribute__((ext_vector_type(16))) __bf16 v16bf;
typedef __attribute__((ext_vector_type(8)))  __bf16 v8bf;
typedef __attribute__((ext_vector_type(4)))  __bf16 v4bf;
typedef __attribute__((ext_vector_type(8)))  float  v8f;
typedef __attribute__((ext_vector_type(4)))  float  v4f;
typedef __attribute__((ext_vector_type(4)))  int    v4i;

#define B_    8
#define C_    512
#define N_    1024
#define HEADS 8
#define HD    64
#define G_    32
#define CPG   16   // channels per group

#if defined(__HIP_DEVICE_COMPILE__) && __has_builtin(__builtin_amdgcn_global_load_async_to_lds_b128)
#define USE_ASYNC_LDS 1
#else
#define USE_ASYNC_LDS 0
#endif

// float -> bf16 round-to-nearest-even via bit ops
static __device__ __forceinline__ __bf16 f2bf(float f) {
  union { float f; unsigned u; } v; v.f = f;
  unsigned r = (v.u + 0x7FFFu + ((v.u >> 16) & 1u)) >> 16;
  unsigned short h = (unsigned short)r;
  return __builtin_bit_cast(__bf16, h);
}

static __device__ __forceinline__ v16bf concat8(v8bf lo, v8bf hi) {
  return __builtin_shufflevector(lo, hi, 0, 1, 2, 3, 4, 5, 6, 7,
                                 8, 9, 10, 11, 12, 13, 14, 15);
}

// 16-byte global->LDS copy: async (no VGPR round-trip) when available.
static __device__ __forceinline__ void copy16_g2l(void* lds, const void* g) {
#if USE_ASYNC_LDS
  __builtin_amdgcn_global_load_async_to_lds_b128(
      (__attribute__((address_space(1))) v4i*)(g),
      (__attribute__((address_space(3))) v4i*)(lds), 0, 0);
#else
  *(v8bf*)lds = *(const v8bf*)g;
#endif
}

static __device__ __forceinline__ void g2l_wait() {
#if USE_ASYNC_LDS
#if __has_builtin(__builtin_amdgcn_s_wait_asynccnt)
  __builtin_amdgcn_s_wait_asynccnt(0);
#else
  asm volatile("s_wait_asynccnt 0x0" ::: "memory");
#endif
#endif
}

// ---------------------------------------------------------------------------
// Kernel 0: one-time fp32 -> bf16 weight conversion (keeps conversions out of
// the GEMM K-loop).  n8 = element count / 8.
// ---------------------------------------------------------------------------
__global__ void convert_bf16_kernel(const float* __restrict__ src,
                                    __bf16* __restrict__ dst, int n8) {
  int i = blockIdx.x * blockDim.x + threadIdx.x;
  if (i < n8) {
    v4f f0 = *(const v4f*)&src[(size_t)i * 8];
    v4f f1 = *(const v4f*)&src[(size_t)i * 8 + 4];
    v8bf w;
    #pragma unroll
    for (int j = 0; j < 4; ++j) { w[j] = f2bf(f0[j]); w[j + 4] = f2bf(f1[j]); }
    *(v8bf*)&dst[(size_t)i * 8] = w;
  }
}

// ---------------------------------------------------------------------------
// Kernel 1: GroupNorm -> bf16 x_norm.  One block per (batch, group).
// ---------------------------------------------------------------------------
__global__ void groupnorm_kernel(const float* __restrict__ x,
                                 const float* __restrict__ gamma,
                                 const float* __restrict__ beta,
                                 __bf16* __restrict__ xnorm) {
  __shared__ float s_sum[256];
  __shared__ float s_ss[256];
  const int tid = threadIdx.x;
  const int b = blockIdx.x >> 5;
  const int g = blockIdx.x & 31;
  const float* xp = x + (size_t)b * C_ * N_ + (size_t)g * CPG * N_;
  const v4f* xp4 = (const v4f*)xp;

  float sum = 0.f, ss = 0.f;
  #pragma unroll 4
  for (int t = 0; t < 16; ++t) {          // 4096 float4 / 256 threads
    v4f v = xp4[tid + t * 256];
    #pragma unroll
    for (int j = 0; j < 4; ++j) { sum += v[j]; ss += v[j] * v[j]; }
  }
  s_sum[tid] = sum; s_ss[tid] = ss;
  __syncthreads();
  for (int off = 128; off > 0; off >>= 1) {
    if (tid < off) { s_sum[tid] += s_sum[tid + off]; s_ss[tid] += s_ss[tid + off]; }
    __syncthreads();
  }
  const float mean = s_sum[0] * (1.f / 16384.f);
  const float var  = s_ss[0] * (1.f / 16384.f) - mean * mean;
  const float rstd = rsqrtf(var + 1e-5f);

  __bf16* op = xnorm + (size_t)b * C_ * N_ + (size_t)g * CPG * N_;
  #pragma unroll 4
  for (int t = 0; t < 16; ++t) {
    int idx4 = tid + t * 256;
    int c = idx4 >> 8;                    // 256 float4 per channel row
    v4f v = xp4[idx4];
    float ga = gamma[g * CPG + c], be = beta[g * CPG + c];
    v4bf pk;
    #pragma unroll
    for (int j = 0; j < 4; ++j) pk[j] = f2bf((v[j] - mean) * rstd * ga + be);
    *(v4bf*)&op[idx4 * 4] = pk;
  }
}

// ---------------------------------------------------------------------------
// Kernel 2/4: out[b] = Wbf[M,K] @ X[b][K,1024] + bias, via WMMA bf16.
// Block tile 64x64, K tiled by 32; wave w -> rows 16*(w&3), cols 32*(w>>2).
// W stage: straight async b128 copies.  X stage: transposed to Xs[n][k].
// mode 0: bf16 store (QKV).  mode 1: fp32 store with residual add (proj).
// ---------------------------------------------------------------------------
__global__ void gemm_wmma_kernel(const __bf16* __restrict__ Wbf,
                                 const __bf16* __restrict__ X,
                                 const float* __restrict__ bias,
                                 __bf16* __restrict__ outb,
                                 float* __restrict__ outf,
                                 const float* __restrict__ resid,
                                 int M, int K, int mode) {
  __shared__ __align__(16) __bf16 Ws[64][40];   // [m][k]
  __shared__ __align__(16) __bf16 Xs[64][40];   // [n][k] (transposed stage)
  const int tid  = threadIdx.x;
  const int lane = tid & 31, wv = tid >> 5;
  const int half = lane >> 4, l15 = lane & 15;
  const int rg = 16 * (wv & 3);
  const int cg = 32 * (wv >> 2);
  const int m0 = blockIdx.x * 64, n0 = blockIdx.y * 64, b = blockIdx.z;
  const __bf16* Xb = X + (size_t)b * K * N_;

  v8f acc0 = {}, acc1 = {};
  const int nk = K >> 5;
  for (int kt = 0; kt < nk; ++kt) {
    const int k0 = kt * 32;
    // stage W tile 64x32 bf16: straight 16B copies (async path)
    {
      int i = tid >> 2, jv = (tid & 3) * 8;
      copy16_g2l(&Ws[i][jv], &Wbf[(size_t)(m0 + i) * K + k0 + jv]);
    }
    // stage X tile 32k x 64n, transposed to Xs[n][k]
    {
      int k = tid >> 3, nv = (tid & 7) * 8;
      v8bf xv = *(const v8bf*)&Xb[(size_t)(k0 + k) * N_ + n0 + nv];
      #pragma unroll
      for (int j = 0; j < 8; ++j) Xs[nv + j][k] = xv[j];
    }
    if (kt + 1 < nk) {
      __builtin_prefetch(&Wbf[(size_t)(m0 + (tid >> 2)) * K + k0 + 32], 0, 1);
      __builtin_prefetch(&Xb[(size_t)(k0 + 32 + (tid >> 3)) * N_ + n0], 0, 1);
    }
    g2l_wait();
    __syncthreads();

    // A: lane row rg+l15, K run [16h..16h+15] -> 2x b128
    const __bf16* ar = &Ws[rg + l15][0];
    v16bf a = concat8(*(const v8bf*)(ar + 16 * half),
                      *(const v8bf*)(ar + 16 * half + 8));
    // B: lane col, K run [16h..16h+15] -> 2x b128 each
    const __bf16* b0r = &Xs[cg + l15][16 * half];
    const __bf16* b1r = &Xs[cg + 16 + l15][16 * half];
    v16bf bb0 = concat8(*(const v8bf*)b0r, *(const v8bf*)(b0r + 8));
    v16bf bb1 = concat8(*(const v8bf*)b1r, *(const v8bf*)(b1r + 8));
    acc0 = __builtin_amdgcn_wmma_f32_16x16x32_bf16(false, a, false, bb0,
                                                   (short)0, acc0, false, false);
    acc1 = __builtin_amdgcn_wmma_f32_16x16x32_bf16(false, a, false, bb1,
                                                   (short)0, acc1, false, false);
    __syncthreads();
  }

  // Epilogue. D layout: elem r -> row M = r + 8*half, col N = l15.
  #pragma unroll
  for (int r = 0; r < 8; ++r) {
    int m = m0 + rg + r + 8 * half;
    int n = n0 + cg + l15;
    float v0 = acc0[r] + bias[m];
    float v1 = acc1[r] + bias[m];
    if (mode == 0) {
      size_t o = (size_t)b * M * N_ + (size_t)m * N_;
      outb[o + n]      = f2bf(v0);
      outb[o + n + 16] = f2bf(v1);
    } else {
      size_t o = (size_t)b * C_ * N_ + (size_t)m * N_;
      outf[o + n]      = resid[o + n] + v0;
      outf[o + n + 16] = resid[o + n + 16] + v1;
    }
  }
}

// ---------------------------------------------------------------------------
// Kernel 3: flash attention.  Block = (64 query rows, head, batch).
// S = Q^T K (scaled), online softmax, O += P @ V^T.
// LDS layouts (operand-major, all gathers are b128):
//   Qs[n][c] (A for S), Kst[m][c] (B for S), Pbuf[n][m] (A for O),
//   Vst[c][m] (B for O, staged via async b128 copies).
// ---------------------------------------------------------------------------
__global__ void attention_kernel(const __bf16* __restrict__ qkv,
                                 __bf16* __restrict__ attnout) {
  __shared__ __align__(16) __bf16 Qs[64][72];    // [n][c]
  __shared__ __align__(16) __bf16 Kst[32][72];   // [m][c]
  __shared__ __align__(16) __bf16 Vst[64][40];   // [c][m]
  __shared__ float  Sbuf[64][33];
  __shared__ __align__(16) __bf16 Pbuf[64][40];  // [n][m]
  __shared__ float rowMax[64], rowSum[64], resc[64];

  const int tid  = threadIdx.x;
  const int lane = tid & 31, wv = tid >> 5;
  const int half = lane >> 4, l15 = lane & 15;
  const int rg = 16 * (wv & 3);    // query-row group
  const int cg = 16 * (wv >> 2);   // S col group
  const int vg = 32 * (wv >> 2);   // O col group
  const int n0 = blockIdx.x * 64;
  const int h = blockIdx.y, b = blockIdx.z;

  const __bf16* qb = qkv + (size_t)b * 3 * C_ * N_ + (size_t)(h * HD) * N_;
  const __bf16* kb = qb + (size_t)C_ * N_;
  const __bf16* vb = qb + (size_t)(2 * C_) * N_;

  // Stage Q^T once: Qs[n][c] = q[c][n0+n]  (b128 global loads, b16 scatter)
  #pragma unroll
  for (int t = 0; t < 2; ++t) {
    int vecid = tid + t * 256;          // 512 vectors of 8
    int c = vecid >> 3, nv = (vecid & 7) * 8;
    v8bf qv = *(const v8bf*)&qb[(size_t)c * N_ + n0 + nv];
    #pragma unroll
    for (int j = 0; j < 8; ++j) Qs[nv + j][c] = qv[j];
  }
  if (tid < 64) { rowMax[tid] = -1e30f; rowSum[tid] = 0.f; }
  v8f o0 = {}, o1 = {};
  __syncthreads();

  for (int mt = 0; mt < 32; ++mt) {
    const int m0 = mt * 32;
    // Vst[c][m] = v[c][m0+m] (straight async b128 copies)
    {
      int c = tid >> 2, mv = (tid & 3) * 8;
      copy16_g2l(&Vst[c][mv], &vb[(size_t)c * N_ + m0 + mv]);
    }
    // Kst[m][c] = k[c][m0+m] (transpose scatter)
    {
      int c = tid >> 2, mv = (tid & 3) * 8;
      v8bf kv = *(const v8bf*)&kb[(size_t)c * N_ + m0 + mv];
      #pragma unroll
      for (int j = 0; j < 8; ++j) Kst[mv + j][c] = kv[j];
    }
    g2l_wait();
    __syncthreads();

    // ---- S tile (64x32): per-wave 16x16, K=64 via two 16x16x32 WMMAs ----
    v8f sacc = {};
    #pragma unroll
    for (int kk = 0; kk < 2; ++kk) {
      const __bf16* ar = &Qs[rg + l15][32 * kk];
      v16bf a = concat8(*(const v8bf*)(ar + 16 * half),
                        *(const v8bf*)(ar + 16 * half + 8));
      const __bf16* br = &Kst[cg + l15][32 * kk + 16 * half];
      v16bf bb = concat8(*(const v8bf*)br, *(const v8bf*)(br + 8));
      sacc = __builtin_amdgcn_wmma_f32_16x16x32_bf16(false, a, false, bb,
                                                     (short)0, sacc, false, false);
    }
    #pragma unroll
    for (int r = 0; r < 8; ++r)
      Sbuf[rg + r + 8 * half][cg + l15] = sacc[r] * 0.125f;   // hd^-0.5
    __syncthreads();

    // ---- online softmax, 64 rows by first 64 lanes ----
    if (tid < 64) {
      int row = tid;
      float mo = rowMax[row];
      float mn = mo;
      #pragma unroll 8
      for (int j = 0; j < 32; ++j) mn = fmaxf(mn, Sbuf[row][j]);
      float rf = __expf(mo - mn);
      float s = 0.f;
      #pragma unroll 8
      for (int j = 0; j < 32; ++j) {
        float p = __expf(Sbuf[row][j] - mn);
        Pbuf[row][j] = f2bf(p);
        s += p;
      }
      rowMax[row] = mn;
      rowSum[row] = rowSum[row] * rf + s;
      resc[row] = rf;
    }
    __syncthreads();

    // ---- rescale running O, then O += P @ V^T (K = m tile = 32) ----
    #pragma unroll
    for (int r = 0; r < 8; ++r) {
      float rf = resc[rg + r + 8 * half];
      o0[r] *= rf; o1[r] *= rf;
    }
    {
      const __bf16* ar = &Pbuf[rg + l15][0];
      v16bf a = concat8(*(const v8bf*)(ar + 16 * half),
                        *(const v8bf*)(ar + 16 * half + 8));
      const __bf16* b0r = &Vst[vg + l15][16 * half];
      const __bf16* b1r = &Vst[vg + 16 + l15][16 * half];
      v16bf bb0 = concat8(*(const v8bf*)b0r, *(const v8bf*)(b0r + 8));
      v16bf bb1 = concat8(*(const v8bf*)b1r, *(const v8bf*)(b1r + 8));
      o0 = __builtin_amdgcn_wmma_f32_16x16x32_bf16(false, a, false, bb0,
                                                   (short)0, o0, false, false);
      o1 = __builtin_amdgcn_wmma_f32_16x16x32_bf16(false, a, false, bb1,
                                                   (short)0, o1, false, false);
    }
    __syncthreads();
  }

  // ---- normalize and store O^T as [b][h*64+c][n] bf16 for proj GEMM ----
  __bf16* ob = attnout + (size_t)b * C_ * N_ + (size_t)(h * HD) * N_;
  #pragma unroll
  for (int r = 0; r < 8; ++r) {
    int row = rg + r + 8 * half;
    float inv = 1.f / rowSum[row];
    ob[(size_t)(vg + l15) * N_ + n0 + row]      = f2bf(o0[r] * inv);
    ob[(size_t)(vg + 16 + l15) * N_ + n0 + row] = f2bf(o1[r] * inv);
  }
}

// ---------------------------------------------------------------------------
extern "C" void kernel_launch(void* const* d_in, const int* in_sizes, int n_in,
                              void* d_out, int out_size, void* d_ws, size_t ws_size,
                              hipStream_t stream) {
  const float* x      = (const float*)d_in[0];
  const float* gamma  = (const float*)d_in[1];
  const float* beta   = (const float*)d_in[2];
  const float* w_qkv  = (const float*)d_in[3];
  const float* b_qkv  = (const float*)d_in[4];
  const float* w_proj = (const float*)d_in[5];
  const float* b_proj = (const float*)d_in[6];
  float* out = (float*)d_out;

  // workspace: xnorm bf16 (8 MB) | qkv bf16 (24 MB) | attnout bf16 (8 MB)
  //            | w_qkv bf16 (1.5 MB) | w_proj bf16 (0.5 MB)
  char* ws = (char*)d_ws;
  __bf16* xnorm   = (__bf16*)(ws);
  __bf16* qkv     = (__bf16*)(ws + (size_t)B_ * C_ * N_ * 2);
  __bf16* attnout = (__bf16*)(ws + (size_t)B_ * C_ * N_ * 2
                                 + (size_t)B_ * 3 * C_ * N_ * 2);
  __bf16* wqkv_bf = (__bf16*)(ws + (size_t)B_ * C_ * N_ * 4
                                 + (size_t)B_ * 3 * C_ * N_ * 2);
  __bf16* wproj_bf = wqkv_bf + (size_t)3 * C_ * C_;

  // 0. one-time weight conversion to bf16
  convert_bf16_kernel<<<dim3((3 * C_ * C_ / 8 + 255) / 256), dim3(256), 0, stream>>>(
      w_qkv, wqkv_bf, 3 * C_ * C_ / 8);
  convert_bf16_kernel<<<dim3((C_ * C_ / 8 + 255) / 256), dim3(256), 0, stream>>>(
      w_proj, wproj_bf, C_ * C_ / 8);

  // 1. GroupNorm -> bf16
  groupnorm_kernel<<<dim3(B_ * G_), dim3(256), 0, stream>>>(x, gamma, beta, xnorm);

  // 2. QKV GEMM: [1536,512] x [512,1024] per batch
  gemm_wmma_kernel<<<dim3(3 * C_ / 64, N_ / 64, B_), dim3(256), 0, stream>>>(
      wqkv_bf, xnorm, b_qkv, qkv, nullptr, nullptr, 3 * C_, C_, 0);

  // 3. Flash attention per (64-query chunk, head, batch)
  attention_kernel<<<dim3(N_ / 64, HEADS, B_), dim3(256), 0, stream>>>(qkv, attnout);

  // 4. Proj GEMM + bias + residual -> fp32 out
  gemm_wmma_kernel<<<dim3(C_ / 64, N_ / 64, B_), dim3(256), 0, stream>>>(
      wproj_bf, attnout, b_proj, nullptr, out, x, C_, C_, 1);
}
